// BiLSTMDecoderModel_83013127897322
// MI455X (gfx1250) — compile-verified, matrix operand order checked by reading, not run on hardware
//
#include <hip/hip_runtime.h>
#include <hip/hip_bf16.h>
#include <cstdint>
#include <cstddef>

// ---------------- model dims ----------------
#define CC 6
#define EE 300
#define EP 320       // E padded to a multiple of 32 (WMMA K granularity)
#define HH 256
#define G4 1024      // 4*H (LSTM gates)
#define BB 128
#define SS 512
#define H2 512       // 2*H (decoder hidden)
#define G3 1536      // 3*2H (GRU gates)
#define PP 256
#define KL 576       // EP + HH : combined K for fused [x|h] @ [Wih|Whh]^T
#define NKT_H 16     // H2/32

typedef __attribute__((ext_vector_type(16))) __bf16        v16bf;
typedef __attribute__((ext_vector_type(8)))  float          v8f;
typedef __attribute__((ext_vector_type(8)))  unsigned short ushort8;

__device__ __forceinline__ unsigned short f2bf(float f) {
  unsigned int u = __builtin_bit_cast(unsigned int, f);
  u += 0x7fffu + ((u >> 16) & 1u);          // round to nearest even
  return (unsigned short)(u >> 16);
}
__device__ __forceinline__ float bf2f(unsigned short h) {
  unsigned int u = ((unsigned int)h) << 16;
  return __builtin_bit_cast(float, u);
}
__device__ __forceinline__ float sigm(float x) { return 1.f / (1.f + __expf(-x)); }

// Defeat LICM without destroying pointer provenance: an opaque ZERO offset.
// Adding it to a global pointer keeps addrspace inference (global_load, not
// flat_load) while making addresses non-loop-invariant, so weight fragments
// stream from L2 each step instead of being hoisted+spilled to scratch.
__device__ __forceinline__ size_t opaque_zero() {
  unsigned int z = 0;
  asm volatile("" : "+s"(z));
  return (size_t)z;
}

// Load a bf16 A/B fragment from a row-major [row][k] matrix.
// lane l: row_in_tile = l&15 ; k = kbase + {0..7} and kbase+16+{0..7}, kbase=(l>>4)*8.
// caller passes p = &M[tile*16 + (lane&15)][ktile*32 + (lane>>4)*8]
__device__ __forceinline__ v16bf ldfrag(const unsigned short* p) {
  union { v16bf v; ushort8 h[2]; } u;
  u.h[0] = *(const ushort8*)(p);
  u.h[1] = *(const ushort8*)(p + 16);
  return u.v;
}
__device__ __forceinline__ v8f wmma_bf16(v16bf a, v16bf b, v8f c) {
  return __builtin_amdgcn_wmma_f32_16x16x32_bf16(false, a, false, b, (short)0, c,
                                                 false, false);
}

// ---------------- prep kernels ----------------

// seq_emb = tanh(embed_W[seq]) time-major, bf16, padded to EP. [S][B][EP]
__global__ void k_embed(const int* __restrict__ seq, const float* __restrict__ embW,
                        unsigned short* __restrict__ xemb) {
  int sb = blockIdx.x;             // s*BB + b
  int s = sb >> 7, b = sb & 127;
  int v = seq[(size_t)b * SS + s];
  int e = threadIdx.x;
  float val = (e < EE) ? tanhf(embW[(size_t)v * EE + e]) : 0.f;
  xemb[(size_t)sb * EP + e] = f2bf(val);
}

// combined weight [G4][KL]: cols [0,300)=Wih, [300,320)=0, [320,576)=Whh, bf16
__global__ void k_wcomb(const float* __restrict__ Wih, const float* __restrict__ Whh,
                        unsigned short* __restrict__ out) {
  int idx = blockIdx.x * blockDim.x + threadIdx.x;
  if (idx >= G4 * KL) return;
  int g = idx / KL, k = idx % KL;
  float v = 0.f;
  if (k < EE)            v = Wih[(size_t)g * EE + k];
  else if (k >= EP)      v = Whh[(size_t)g * HH + (k - EP)];
  out[idx] = f2bf(v);
}

__global__ void k_bias2(const float* __restrict__ a, const float* __restrict__ b,
                        float* __restrict__ out, int n) {
  int i = blockIdx.x * blockDim.x + threadIdx.x;
  if (i < n) out[i] = a[i] + b[i];
}

__global__ void k_f2bf(const float* __restrict__ in, unsigned short* __restrict__ out,
                       int n) {
  int i = blockIdx.x * blockDim.x + threadIdx.x;
  if (i < n) out[i] = f2bf(in[i]);
}

// gx_dec[c][g] = tanh(embed_class_W[classes[c]]) . d_Wih[g] + d_bih[g]   (fp32, tiny)
__global__ void k_gxdec(const int* __restrict__ classes, const float* __restrict__ ecW,
                        const float* __restrict__ dWih, const float* __restrict__ dbih,
                        float* __restrict__ gxdec) {
  int idx = blockIdx.x * blockDim.x + threadIdx.x;
  if (idx >= CC * G3) return;
  int c = idx / G3, g = idx % G3;
  int cl = classes[c];
  float acc = dbih[g];
  for (int e = 0; e < EE; ++e)
    acc += tanhf(ecW[(size_t)cl * EE + e]) * dWih[(size_t)g * EE + e];
  gxdec[idx] = acc;
}

// ---------------- BiLSTM recurrence ----------------
// grid = 8 blocks: blockIdx = dir*4 + batch_chunk (32 rows each).
// 512 thr = 16 waves: wave w owns h-column tile j=w and ALL FOUR gate ntiles for it
// (i,f,g,o), so the c/h update runs entirely on accumulator registers.
// Per-wave live state: 2 Mtiles x 4 gates = 8 v8f acc (64 VGPR) + c-state (16)
// + in-flight A/B fragments => < 256 VGPRs, no spills.
// h lives in LDS (bf16 [32][256]); W streams from L2 (1.18 MB, resident) every
// step — the opaque_zero() offset keeps the loads un-hoistable yet global.
__global__ __launch_bounds__(512) void k_lstm(
    const unsigned short* __restrict__ xemb,
    const unsigned short* __restrict__ Wf, const unsigned short* __restrict__ Wb,
    const float* __restrict__ bcf, const float* __restrict__ bcb,
    float* __restrict__ out0) {
  const int dir = blockIdx.x >> 2, chunk = blockIdx.x & 3;
  const unsigned short* W = dir ? Wb : Wf;
  const float* bc = dir ? bcb : bcf;

  __shared__ __align__(16) unsigned short hl[32 * HH];   // 16 KB

  const int tid = threadIdx.x, lane = tid & 31, wave = tid >> 5;
  const int lm = lane & 15, lk8 = (lane >> 4) * 8;
  const int j = wave;                       // h-column tile 0..15

  for (int i = tid; i < 32 * HH; i += 512) hl[i] = 0;

  float bias[4];
#pragma unroll
  for (int gt = 0; gt < 4; ++gt)
    bias[gt] = bc[(gt * 16 + j) * 16 + lm];

  // per-lane offsets (constant over the time loop)
  const size_t xoff0 = (size_t)(chunk * 32 + 0  + lm) * EP + lk8;
  const size_t xoff1 = (size_t)(chunk * 32 + 16 + lm) * EP + lk8;
  const unsigned short* hlb0 = hl + (size_t)(0  + lm) * HH + lk8;
  const unsigned short* hlb1 = hl + (size_t)(16 + lm) * HH + lk8;
  size_t woff[4];
#pragma unroll
  for (int gt = 0; gt < 4; ++gt)
    woff[gt] = (size_t)((gt * 16 + j) * 16 + lm) * KL + lk8;

  v8f cacc[2];
#pragma unroll
  for (int mt = 0; mt < 2; ++mt)
#pragma unroll
    for (int e = 0; e < 8; ++e) cacc[mt][e] = 0.f;

  __syncthreads();

  for (int s = 0; s < SS; ++s) {
    // backward scan reads seq_emb[(-i) % S] = {0, S-1, ..., 1}
    const int sx = dir ? ((s == 0) ? 0 : (SS - s)) : s;
    const unsigned short* xrow = xemb + (size_t)sx * BB * EP;
    const unsigned short* Wv = W + opaque_zero();  // re-stream from L2 each step

    v8f acc[2][4];
#pragma unroll
    for (int mt = 0; mt < 2; ++mt)
#pragma unroll
      for (int gt = 0; gt < 4; ++gt) {
        const float bv = bias[gt];
#pragma unroll
        for (int e = 0; e < 8; ++e) acc[mt][gt][e] = bv;
      }

    // --- phase 1: input contribution, K = 0..320 (embeddings from global/L2)
    for (int kt = 0; kt < EP / 32; ++kt) {
      const int kk = kt * 32;
      v16bf a0 = ldfrag(xrow + xoff0 + kk);
      v16bf a1 = ldfrag(xrow + xoff1 + kk);
#pragma unroll
      for (int gt = 0; gt < 4; ++gt) {
        v16bf bf = ldfrag(Wv + woff[gt] + kk);
        acc[0][gt] = wmma_bf16(a0, bf, acc[0][gt]);
        acc[1][gt] = wmma_bf16(a1, bf, acc[1][gt]);
      }
    }
    // --- phase 2: hidden contribution, K = 320..576 (h from LDS)
    for (int kt = 0; kt < HH / 32; ++kt) {
      const int kk = kt * 32;
      v16bf a0 = ldfrag(hlb0 + kk);
      v16bf a1 = ldfrag(hlb1 + kk);
#pragma unroll
      for (int gt = 0; gt < 4; ++gt) {
        v16bf bf = ldfrag(Wv + woff[gt] + EP + kk);
        acc[0][gt] = wmma_bf16(a0, bf, acc[0][gt]);
        acc[1][gt] = wmma_bf16(a1, bf, acc[1][gt]);
      }
    }

    __syncthreads();   // all waves done reading old h

    const int col = j * 16 + lm;
#pragma unroll
    for (int mt = 0; mt < 2; ++mt) {
#pragma unroll
      for (int e = 0; e < 8; ++e) {
        float c = sigm(acc[mt][1][e]) * cacc[mt][e] +
                  sigm(acc[mt][0][e]) * tanhf(acc[mt][2][e]);
        float h = sigm(acc[mt][3][e]) * tanhf(c);
        cacc[mt][e] = c;
        const int ml = mt * 16 + lk8 + e;        // local batch row 0..31
        hl[ml * HH + col] = f2bf(h);
        if (s == SS - 1)
          out0[(size_t)(chunk * 32 + ml) * H2 + dir * HH + col] = h;
      }
    }
    __syncthreads();
  }
}

// ---------------- GRU decoder + proj + cls + log_softmax ----------------
// 1 block, 512 thr = 16 waves. h double-buffered (bf16) in L2-resident global ws.
// Each wave owns j-tiles {2w, 2w+1}, processed sequentially (3 live accumulators).
__global__ __launch_bounds__(512) void k_dec(
    const float* __restrict__ out0, const float* __restrict__ gxdec,
    const unsigned short* __restrict__ dWhhB, const float* __restrict__ dbhh,
    const unsigned short* __restrict__ pWB, const float* __restrict__ pb,
    const float* __restrict__ clsW, const float* __restrict__ clsb,
    unsigned short* __restrict__ hbuf, float* __restrict__ projbuf,
    float* __restrict__ out) {
  const int tid = threadIdx.x, lane = tid & 31, wave = tid >> 5;
  const int lm = lane & 15, lk8 = (lane >> 4) * 8;
  const int j0 = wave * 2;

  for (int i = tid; i < BB * H2; i += 512) hbuf[i] = f2bf(out0[i]);
  __threadfence();
  __syncthreads();

  for (int c = 0; c < CC; ++c) {
    const unsigned short* hcur0 = hbuf + (size_t)(c & 1) * BB * H2;
    unsigned short* hnxt = hbuf + (size_t)((c + 1) & 1) * BB * H2;
    const float* gx = gxdec + (size_t)c * G3;

    for (int mt = 0; mt < 8; ++mt) {     // B = 128 -> 8 M tiles
#pragma unroll 1
      for (int jj = 0; jj < 2; ++jj) {   // sequential: keeps only 3 acc live
        const int jt = j0 + jj;
        const unsigned short* hcur = hcur0 + opaque_zero();
        const unsigned short* dW = dWhhB + opaque_zero();
        v8f acc[3];
#pragma unroll
        for (int gt = 0; gt < 3; ++gt) {
          const float bv = dbhh[gt * H2 + jt * 16 + lm];
#pragma unroll
          for (int e = 0; e < 8; ++e) acc[gt][e] = bv;
        }
        for (int kt = 0; kt < NKT_H; ++kt) {
          const int kk = kt * 32 + lk8;
          v16bf a = ldfrag(hcur + (size_t)(mt * 16 + lm) * H2 + kk);
#pragma unroll
          for (int gt = 0; gt < 3; ++gt) {
            const int nt = gt * 32 + jt;   // r,z,n blocks of 512 cols each
            v16bf bf = ldfrag(dW + (size_t)(nt * 16 + lm) * H2 + kk);
            acc[gt] = wmma_bf16(a, bf, acc[gt]);
          }
        }
        const int col = jt * 16 + lm;
        const float gxr = gx[0 * H2 + col];
        const float gxz = gx[1 * H2 + col];
        const float gxn = gx[2 * H2 + col];
#pragma unroll
        for (int e = 0; e < 8; ++e) {
          const int m = mt * 16 + lk8 + e;
          float r = sigm(gxr + acc[0][e]);
          float z = sigm(gxz + acc[1][e]);
          float n = tanhf(gxn + r * acc[2][e]);
          float hp = bf2f(hcur[(size_t)m * H2 + col]);
          float hn = tanhf((1.f - z) * n + z * hp);
          hnxt[(size_t)m * H2 + col] = f2bf(hn);
        }
      }
    }
    __threadfence();
    __syncthreads();

    // proj = h_new @ proj_W^T + proj_b  -> projbuf [128][256] (WMMA)
#pragma unroll 1
    for (int i = 0; i < 8; ++i) {
      const int p = wave * 8 + i;        // 0..127 (m,n) tile pairs
      const int mt = p >> 4, nt = p & 15;
      const unsigned short* hv = (const unsigned short*)hnxt + opaque_zero();
      const unsigned short* pW = pWB + opaque_zero();
      v8f acc;
      const float pbv = pb[nt * 16 + lm];
#pragma unroll
      for (int e = 0; e < 8; ++e) acc[e] = pbv;
      for (int kt = 0; kt < NKT_H; ++kt) {
        const int kk = kt * 32 + lk8;
        v16bf a = ldfrag(hv + (size_t)(mt * 16 + lm) * H2 + kk);
        v16bf bf = ldfrag(pW + (size_t)(nt * 16 + lm) * H2 + kk);
        acc = wmma_bf16(a, bf, acc);
      }
#pragma unroll
      for (int e = 0; e < 8; ++e) {
        const int m = mt * 16 + lk8 + e;
        projbuf[(size_t)m * PP + nt * 16 + lm] = acc[e];
      }
    }
    __threadfence();
    __syncthreads();

    // logits + log_softmax (tiny)
    if (tid < BB) {
      const int b = tid;
      float l0 = clsb[0], l1 = clsb[1];
      for (int k = 0; k < PP; ++k) {
        const float pv = projbuf[(size_t)b * PP + k];
        l0 += pv * clsW[k];
        l1 += pv * clsW[PP + k];
      }
      const float mx = fmaxf(l0, l1);
      const float lse = mx + logf(__expf(l0 - mx) + __expf(l1 - mx));
      out[((size_t)c * BB + b) * 2 + 0] = l0 - lse;
      out[((size_t)c * BB + b) * 2 + 1] = l1 - lse;
    }
    __threadfence();
    __syncthreads();
  }
}

// ---------------- launch ----------------
extern "C" void kernel_launch(void* const* d_in, const int* in_sizes, int n_in,
                              void* d_out, int out_size, void* d_ws, size_t ws_size,
                              hipStream_t stream) {
  const int*   seq   = (const int*)d_in[0];
  const int*   clss  = (const int*)d_in[1];
  const float* embW  = (const float*)d_in[2];
  const float* ecW   = (const float*)d_in[3];
  const float* fWih  = (const float*)d_in[4];
  const float* fWhh  = (const float*)d_in[5];
  const float* fbih  = (const float*)d_in[6];
  const float* fbhh  = (const float*)d_in[7];
  const float* bWih  = (const float*)d_in[8];
  const float* bWhh  = (const float*)d_in[9];
  const float* bbih  = (const float*)d_in[10];
  const float* bbhh  = (const float*)d_in[11];
  const float* dWih  = (const float*)d_in[12];
  const float* dWhh  = (const float*)d_in[13];
  const float* dbih  = (const float*)d_in[14];
  const float* dbhh  = (const float*)d_in[15];
  const float* projW = (const float*)d_in[16];
  const float* projb = (const float*)d_in[17];
  const float* clsW  = (const float*)d_in[18];
  const float* clsb  = (const float*)d_in[19];

  char* ws = (char*)d_ws;
  size_t o = 0;
  unsigned short* xemb = (unsigned short*)(ws + o); o += (size_t)SS * BB * EP * 2;  // 41.9 MB
  unsigned short* wcf  = (unsigned short*)(ws + o); o += (size_t)G4 * KL * 2;       // 1.18 MB
  unsigned short* wcb  = (unsigned short*)(ws + o); o += (size_t)G4 * KL * 2;
  float*          bcf  = (float*)(ws + o);          o += (size_t)G4 * 4;
  float*          bcb  = (float*)(ws + o);          o += (size_t)G4 * 4;
  unsigned short* dwhh = (unsigned short*)(ws + o); o += (size_t)G3 * H2 * 2;       // 1.57 MB
  unsigned short* pwbf = (unsigned short*)(ws + o); o += (size_t)PP * H2 * 2;
  float*          gxd  = (float*)(ws + o);          o += (size_t)CC * G3 * 4;
  float*          out0 = (float*)(ws + o);          o += (size_t)BB * H2 * 4;
  unsigned short* hbuf = (unsigned short*)(ws + o); o += (size_t)2 * BB * H2 * 2;
  float*          prjb = (float*)(ws + o);          o += (size_t)BB * PP * 4;
  (void)ws_size; (void)in_sizes; (void)n_in; (void)out_size;

  k_embed<<<SS * BB, EP, 0, stream>>>(seq, embW, xemb);
  k_wcomb<<<(G4 * KL + 255) / 256, 256, 0, stream>>>(fWih, fWhh, wcf);
  k_wcomb<<<(G4 * KL + 255) / 256, 256, 0, stream>>>(bWih, bWhh, wcb);
  k_bias2<<<(G4 + 255) / 256, 256, 0, stream>>>(fbih, fbhh, bcf, G4);
  k_bias2<<<(G4 + 255) / 256, 256, 0, stream>>>(bbih, bbhh, bcb, G4);
  k_f2bf<<<(G3 * H2 + 255) / 256, 256, 0, stream>>>(dWhh, dwhh, G3 * H2);
  k_f2bf<<<(PP * H2 + 255) / 256, 256, 0, stream>>>(projW, pwbf, PP * H2);
  k_gxdec<<<(CC * G3 + 255) / 256, 256, 0, stream>>>(clss, ecW, dWih, dbih, gxd);

  k_lstm<<<8, 512, 0, stream>>>(xemb, wcf, wcb, bcf, bcb, out0);
  k_dec<<<1, 512, 0, stream>>>(out0, gxd, dwhh, dbhh, pwbf, projb, clsW, clsb,
                               hbuf, prjb, (float*)d_out);
}
